// StoLinear_Int_70205535421149
// MI455X (gfx1250) — matrix operand order repaired and checked by reading, not compiled, not run on hardware
//
#include <hip/hip_runtime.h>

typedef __attribute__((ext_vector_type(16))) _Float16 v16h;
typedef __attribute__((ext_vector_type(8)))  float    v8f;

#define IN_F   256
#define OUT_F  128
#define MT     2            // M-tiles per wave (amortize B over M)
#define BLK_M  (8 * 16 * MT)  // 8 waves x 32 rows = 256
#define KSTEPS (IN_F / 32)  // 8
#define NTILES (OUT_F / 16) // 8

// PCG-style hash -> uniform float in [0,1)
__device__ __forceinline__ float uniform_hash(unsigned idx) {
    unsigned s = idx * 747796405u + 2891336453u;
    unsigned w = ((s >> ((s >> 28u) + 4u)) ^ s) * 277803737u;
    w = (w >> 22u) ^ w;
    return (float)(w >> 8) * (1.0f / 16777216.0f);
}

// Build the 16-bit A-matrix (16x32) lane fragment from two 8-float runs 16 apart:
// elems 0-7 = x[row, kb..kb+7], elems 8-15 = x[row, kb+16..kb+23]
__device__ __forceinline__ v16h load_a_frag(const float* __restrict__ xrow,
                                            int ks, int kb) {
    const float4* p0 = (const float4*)(xrow + ks * 32 + kb);
    const float4* p1 = (const float4*)(xrow + ks * 32 + kb + 16);
    float4 a0 = p0[0], a1 = p0[1];
    float4 b0 = p1[0], b1 = p1[1];
    v16h A;
    A[0]  = (_Float16)a0.x; A[1]  = (_Float16)a0.y;
    A[2]  = (_Float16)a0.z; A[3]  = (_Float16)a0.w;
    A[4]  = (_Float16)a1.x; A[5]  = (_Float16)a1.y;
    A[6]  = (_Float16)a1.z; A[7]  = (_Float16)a1.w;
    A[8]  = (_Float16)b0.x; A[9]  = (_Float16)b0.y;
    A[10] = (_Float16)b0.z; A[11] = (_Float16)b0.w;
    A[12] = (_Float16)b1.x; A[13] = (_Float16)b1.y;
    A[14] = (_Float16)b1.z; A[15] = (_Float16)b1.w;
    return A;
}

__global__ __launch_bounds__(256)
void sto_linear_wmma_kernel(const float* __restrict__ x,
                            const int*   __restrict__ weight,
                            const int*   __restrict__ bias,
                            float*       __restrict__ out,
                            int batch) {
    // Dynamic LDS: dequantized weights pre-swizzled into B-matrix wave32 layout.
    //   wqB[((ks*NTILES + nt)*32 + lane)*16 + j] holds Wq[k][n] with
    //   k = ks*32 + (lane>=16)*8 + (j&7) + (j>>3)*16,  n = nt*16 + (lane&15)
    extern __shared__ _Float16 wqB[];

    const int tid = threadIdx.x;

    // --- cooperative dequant + swizzle of W (coalesced global reads) ---
    for (int i = tid; i < IN_F * OUT_F; i += 256) {
        int k   = i >> 7;          // / OUT_F
        int n   = i & (OUT_F - 1);
        int ks  = k >> 5;
        int kin = k & 31;
        int half = (kin >> 3) & 1; // 0: K 0-7,16-23 -> lanes 0-15 ; 1: K 8-15,24-31 -> lanes 16-31
        int j    = (kin & 7) + ((kin >> 4) << 3);
        int nt   = n >> 4;
        int nl   = n & 15;
        int lane_ = half * 16 + nl;
        int dst = ((ks * NTILES + nt) * 32 + lane_) * 16 + j;
        wqB[dst] = (_Float16)(weight[i] >> 5);   // floor(w/32), w >= 0
    }
    __syncthreads();

    const int wave = tid >> 5;
    const int lane = tid & 31;
    const int row0 = blockIdx.x * BLK_M + wave * (16 * MT);  // wave's 32-row stripe
    if (row0 >= batch) return;

    const int kb = (lane >> 4) * 8;                // A half-lane K offset
    const float* __restrict__ xrow0 = x + (size_t)(row0 +      (lane & 15)) * IN_F;
    const float* __restrict__ xrow1 = x + (size_t)(row0 + 16 + (lane & 15)) * IN_F;

    v8f acc[MT][NTILES];
#pragma unroll
    for (int mt = 0; mt < MT; ++mt)
#pragma unroll
        for (int nt = 0; nt < NTILES; ++nt) acc[mt][nt] = (v8f){};

#pragma unroll
    for (int ks = 0; ks < KSTEPS; ++ks) {
        const v16h A0 = load_a_frag(xrow0, ks, kb);
        const v16h A1 = load_a_frag(xrow1, ks, kb);
#pragma unroll
        for (int nt = 0; nt < NTILES; ++nt) {
            const v16h B = ((const v16h*)wqB)[(ks * NTILES + nt) * 32 + lane];
            acc[0][nt] = __builtin_amdgcn_wmma_f32_16x16x32_f16(
                false, A0, false, B, (short)0, acc[0][nt], false, false);
            acc[1][nt] = __builtin_amdgcn_wmma_f32_16x16x32_f16(
                false, A1, false, B, (short)0, acc[1][nt], false, false);
        }
    }

    // --- epilogue: int truncate + quantized bias + sigmoid + Bernoulli ---
    const int ncol = lane & 15;
#pragma unroll
    for (int mt = 0; mt < MT; ++mt) {
        const int mbase = row0 + mt * 16 + (lane >> 4) * 8;  // lanes 16-31 -> rows +8
#pragma unroll
        for (int nt = 0; nt < NTILES; ++nt) {
            const int n  = nt * 16 + ncol;
            const int bq = bias[n] >> 5;          // floor(b/32), L2-resident
#pragma unroll
            for (int r = 0; r < 8; ++r) {
                const int m  = mbase + r;
                const int vi = (int)acc[mt][nt][r] + bq;     // trunc == floor (vals >= 0)
                const float z = (float)vi * 0.015625f;       // / 64
                const float p = 1.0f / (1.0f + __expf(-z));  // sigmoid
                const float u = uniform_hash((unsigned)(m * OUT_F + n) ^ 0x9E3779B9u);
                out[(size_t)m * OUT_F + n] = (u < p) ? 1.0f : 0.0f;
            }
        }
    }
}

extern "C" void kernel_launch(void* const* d_in, const int* in_sizes, int n_in,
                              void* d_out, int out_size, void* d_ws, size_t ws_size,
                              hipStream_t stream) {
    const float* x      = (const float*)d_in[0];
    const int*   weight = (const int*)d_in[1];
    const int*   bias   = (const int*)d_in[2];
    float*       out    = (float*)d_out;

    const int batch = in_sizes[0] / IN_F;               // 65536
    const int grid  = (batch + BLK_M - 1) / BLK_M;      // 256
    const size_t lds_bytes = (size_t)IN_F * OUT_F * sizeof(_Float16); // 64 KB

    sto_linear_wmma_kernel<<<grid, 256, lds_bytes, stream>>>(x, weight, bias, out, batch);
}
